// SparseResBlock_35802847379858
// MI455X (gfx1250) — matrix operand order repaired and checked by reading, not compiled, not run on hardware
//
#include <hip/hip_runtime.h>
#include <stdint.h>

#define NSITES 200000
#define KOFF   27
#define CIN_C  64
#define COUT_C 128

typedef __attribute__((ext_vector_type(16))) _Float16 v16h;
typedef __attribute__((ext_vector_type(8)))  float    v8f;

union FragU { uint4 u[2]; v16h v; };

// ---------------------------------------------------------------------------
// Weight swizzle: f32 W[k][cin][cout] -> f16 B-fragments.
// B-matrix 32x16 f16 layout (ISA 7.12.2 / 7.12.4): lane l holds column
// n = l&15; lanes 0-15 cover K=0..7,16..23, lanes 16-31 cover K=8..15,24..31; VGPR v
// holds pair (K=kk, kk+1), kk = 16*(v>>2) + 8*kh + 2*(v&3). Fragment id f = (k*chunks + q)*8 + t,
// 256 dwords per fragment, lane-major (lane*8 + v).
// ---------------------------------------------------------------------------
__global__ void swizzle_weights(const float* __restrict__ W,
                                uint32_t* __restrict__ dst, int cin)
{
    int chunks = cin >> 5;
    int f = blockIdx.x;
    int t = f & 7;
    int q = (f >> 3) % chunks;
    int k = f / (8 * chunks);
    int lane = threadIdx.x >> 3;   // 0..31
    int v    = threadIdx.x & 7;    // 0..7
    int n  = lane & 15, kh = lane >> 4;
    int kk = (v >> 2) * 16 + kh * 8 + (v & 3) * 2;      // K within 32-chunk
    int ci = q * 32 + kk;
    int co = t * 16 + n;
    const float* s = W + ((size_t)k * cin + ci) * COUT_C + co;
    union { _Float16 h[2]; uint32_t u; } p;
    p.h[0] = (_Float16)s[0];       // K = kk
    p.h[1] = (_Float16)s[COUT_C];  // K = kk+1 (next cin row)
    dst[(size_t)f * 256 + lane * 8 + v] = p.u;
}

__global__ void to_f16(const float* __restrict__ x, _Float16* __restrict__ o,
                       size_t n)
{
    size_t i = (size_t)blockIdx.x * blockDim.x + threadIdx.x;
    if (i < n) o[i] = (_Float16)x[i];
}

__global__ void zero_f32(float* __restrict__ p, int n)
{
    int i = blockIdx.x * blockDim.x + threadIdx.x;
    if (i < n) p[i] = 0.f;
}

// Async global->LDS b128 copy (CDNA5, tracked by ASYNCcnt). LDS offset is the
// low 32 bits of the generic pointer (ISA 10.2).
__device__ __forceinline__ void async_copy_b128(void* lds, const void* gmem)
{
    uint32_t loff  = (uint32_t)(uintptr_t)lds;
    uint64_t gaddr = (uint64_t)(uintptr_t)gmem;
    asm volatile("global_load_async_to_lds_b128 %0, %1, off"
                 :: "v"(loff), "v"(gaddr) : "memory");
}
__device__ __forceinline__ void wait_async0()
{
    asm volatile("s_wait_asynccnt 0x0" ::: "memory");
}

// ---------------------------------------------------------------------------
// Gather + WMMA GEMM. One wave computes a 16x128 output tile with 8 persistent
// f32 accumulators; block = 4 waves = 64 rows. B fragments are staged into
// LDS once per block with double-buffered async global->LDS copies, so the
// weight stream for offset k+1 overlaps the WMMA work on offset k. Also
// accumulates per-channel sum/sumsq for BN (ds_add_f32 -> 1 global atomic).
// A-fragment f16 16x32 layout (ISA 7.12.2): lane half kh covers
// K in {kh*8 .. kh*8+7} U {16+kh*8 .. 23+kh*8}  -> two aligned b128 loads.
// ---------------------------------------------------------------------------
template<int CIN_T, bool GATHER>
__global__ __launch_bounds__(128)
void spconv_wmma(const _Float16* __restrict__ inH,
                 const uint4* __restrict__ wfrag,
                 const int* __restrict__ in_idx,
                 const int* __restrict__ out_idx,
                 float* __restrict__ y,
                 float* __restrict__ gsum, float* __restrict__ gsq,
                 int nk)
{
    constexpr int CHUNKS = CIN_T / 32;
    constexpr int FRAGS  = CHUNKS * 8;          // fragments per offset k
    __shared__ uint4 smemB[2][FRAGS * 64];      // double-buffered k-slices
    __shared__ float ssum[COUT_C];
    __shared__ float ssq[COUT_C];

    int tid = threadIdx.x;
    if (tid < COUT_C) { ssum[tid] = 0.f; ssq[tid] = 0.f; }

    int wave = tid >> 5, lane = tid & 31;
    int m = lane & 15, kh = lane >> 4;
    int i0 = (blockIdx.x * 4 + wave) * 16;

    v8f acc[8] = {};

    // Prologue: stage offset 0's fragments (FRAGS*64 uint4, 128 threads).
    {
        const uint4* g0 = wfrag;
        #pragma unroll
        for (int j = 0; j < FRAGS / 2; ++j) {
            int idx = j * 128 + tid;
            async_copy_b128(&smemB[0][idx], g0 + idx);
        }
    }

    for (int k = 0; k < nk; ++k) {
        wait_async0();          // stage(k) landed in LDS
        __syncthreads();        // visible to all waves; prior readers done

        if (k + 1 < nk) {       // overlap stage(k+1) with compute(k)
            const uint4* gn = wfrag + (size_t)(k + 1) * FRAGS * 64;
            #pragma unroll
            for (int j = 0; j < FRAGS / 2; ++j) {
                int idx = j * 128 + tid;
                async_copy_b128(&smemB[(k + 1) & 1][idx], gn + idx);
            }
            if (k + 2 < nk)     // warm L2 for the stage after that
                __builtin_prefetch(wfrag + (size_t)(k + 2) * FRAGS * 64, 0, 1);
        }

        int row;
        uint32_t mask = 0xFFFFFFFFu;
        if (GATHER) {
            size_t base = (size_t)k * NSITES + i0 + m;
            row  = in_idx[base];
            mask = (out_idx[base] != NSITES) ? 0xFFFFFFFFu : 0u;  // branch-free
        } else {
            row = i0 + m;
        }
        const _Float16* rp = inH + (size_t)row * CIN_T;
        const uint4* wk = smemB[k & 1];

        #pragma unroll
        for (int q = 0; q < CHUNKS; ++q) {
            const uint4* ap =
                reinterpret_cast<const uint4*>(rp + q * 32 + kh * 8);
            uint4 a0 = ap[0];   // K = kh*8 .. kh*8+7
            uint4 a1 = ap[2];   // K = 16+kh*8 .. 23+kh*8  (+32 bytes)
            a0.x &= mask; a0.y &= mask; a0.z &= mask; a0.w &= mask;
            a1.x &= mask; a1.y &= mask; a1.z &= mask; a1.w &= mask;
            FragU A; A.u[0] = a0; A.u[1] = a1;

            const uint4* bq = wk + (size_t)q * 8 * 64 + lane * 2;
            #pragma unroll
            for (int t = 0; t < 8; ++t) {
                FragU B;
                B.u[0] = bq[t * 64];        // ds_load_b128
                B.u[1] = bq[t * 64 + 1];
                acc[t] = __builtin_amdgcn_wmma_f32_16x16x32_f16(
                    false, A.v, false, B.v, (short)0, acc[t], false, false);
            }
        }
    }

    // C/D layout: lane half kh -> M = r + 8*kh, N = lane&15.
    #pragma unroll
    for (int t = 0; t < 8; ++t) {
        int c = t * 16 + m;
        float s = 0.f, s2 = 0.f;
        #pragma unroll
        for (int r = 0; r < 8; ++r) {
            float x = acc[t][r];
            s += x; s2 += x * x;
            y[(size_t)(i0 + kh * 8 + r) * COUT_C + c] = x;
        }
        atomicAdd(&ssum[c], s);   // ds_add_f32
        atomicAdd(&ssq[c], s2);
    }
    __syncthreads();
    if (tid < COUT_C) {
        atomicAdd(&gsum[tid], ssum[tid]);  // global_atomic_add_f32
        atomicAdd(&gsq[tid],  ssq[tid]);
    }
}

// ---------------------------------------------------------------------------
// BN finalize: fold (mean, biased var, gamma, beta) into per-channel a,b.
// ---------------------------------------------------------------------------
__global__ void bn_finalize(const float* __restrict__ gsum,
                            const float* __restrict__ gsq,
                            const float* __restrict__ gamma,
                            const float* __restrict__ beta,
                            float* __restrict__ a, float* __restrict__ b)
{
    int c = threadIdx.x;
    if (c < COUT_C) {
        float inv  = 1.0f / (float)NSITES;
        float mean = gsum[c] * inv;
        float var  = gsq[c] * inv - mean * mean;
        float s    = gamma[c] * rsqrtf(var + 1e-5f);
        a[c] = s;
        b[c] = beta[c] - mean * s;
    }
}

__global__ void bn_relu_f16(const float* __restrict__ y,
                            const float* __restrict__ a,
                            const float* __restrict__ b,
                            _Float16* __restrict__ h)
{
    size_t i = (size_t)blockIdx.x * 256 + threadIdx.x;
    int c = (int)(i & (COUT_C - 1));
    float v = fmaf(y[i], a[c], b[c]);
    h[i] = (_Float16)(v > 0.f ? v : 0.f);
}

__global__ void final_out(const float* __restrict__ y2,
                          const float* __restrict__ a2,
                          const float* __restrict__ b2,
                          const float* __restrict__ ysc,
                          const float* __restrict__ as,
                          const float* __restrict__ bs,
                          float* __restrict__ out)
{
    size_t i = (size_t)blockIdx.x * 256 + threadIdx.x;
    int c = (int)(i & (COUT_C - 1));
    float v = fmaf(y2[i], a2[c], b2[c]) + fmaf(ysc[i], as[c], bs[c]);
    out[i] = v > 0.f ? v : 0.f;
}

// ---------------------------------------------------------------------------
extern "C" void kernel_launch(void* const* d_in, const int* in_sizes, int n_in,
                              void* d_out, int out_size, void* d_ws,
                              size_t ws_size, hipStream_t stream)
{
    (void)in_sizes; (void)n_in; (void)out_size; (void)ws_size;

    const float* feats = (const float*)d_in[0];
    const int*   in_idx  = (const int*)d_in[1];
    const int*   out_idx = (const int*)d_in[2];
    const float* W1  = (const float*)d_in[3];
    const float* g1  = (const float*)d_in[4];
    const float* b1  = (const float*)d_in[5];
    const float* W2  = (const float*)d_in[6];
    const float* g2  = (const float*)d_in[7];
    const float* b2  = (const float*)d_in[8];
    const float* Wsc = (const float*)d_in[9];
    const float* gsc = (const float*)d_in[10];
    const float* bsc = (const float*)d_in[11];
    float* out = (float*)d_out;

    char* ws = (char*)d_ws;
    size_t off = 0;
    auto take = [&](size_t bytes) -> char* {
        char* p = ws + off;
        off += (bytes + 255) & ~(size_t)255;
        return p;
    };

    _Float16* featsH = (_Float16*)take((size_t)NSITES * CIN_C * 2);
    _Float16* hH     = (_Float16*)take((size_t)NSITES * COUT_C * 2);
    float* y1  = (float*)take((size_t)NSITES * COUT_C * 4);  // reused as y2
    float* ysc = (float*)take((size_t)NSITES * COUT_C * 4);
    uint32_t* W1s = (uint32_t*)take((size_t)KOFF * 2 * 8 * 1024);
    uint32_t* W2s = (uint32_t*)take((size_t)KOFF * 4 * 8 * 1024);
    uint32_t* Wss = (uint32_t*)take((size_t)2 * 8 * 1024);
    float* stats = (float*)take(6 * COUT_C * 4);
    float* coeff = (float*)take(6 * COUT_C * 4);
    float *s1 = stats,       *q1 = stats + 128, *s2 = stats + 256,
          *q2 = stats + 384, *sS = stats + 512, *qS = stats + 640;
    float *a1 = coeff,       *c1 = coeff + 128, *a2 = coeff + 256,
          *c2 = coeff + 384, *aS = coeff + 512, *cS = coeff + 640;

    // Prep: zero BN stats, swizzle weights to f16 fragments, feats -> f16.
    zero_f32<<<3, 256, 0, stream>>>(stats, 6 * COUT_C);
    swizzle_weights<<<KOFF * 2 * 8, 256, 0, stream>>>(W1, W1s, CIN_C);
    swizzle_weights<<<KOFF * 4 * 8, 256, 0, stream>>>(W2, W2s, COUT_C);
    swizzle_weights<<<2 * 8, 256, 0, stream>>>(Wsc, Wss, CIN_C);
    to_f16<<<(NSITES * CIN_C + 255) / 256, 256, 0, stream>>>(
        feats, featsH, (size_t)NSITES * CIN_C);

    // conv1 (gather, 27 offsets) and shortcut (1x1, identity gather).
    spconv_wmma<CIN_C, true><<<NSITES / 64, 128, 0, stream>>>(
        featsH, (const uint4*)W1s, in_idx, out_idx, y1, s1, q1, KOFF);
    spconv_wmma<CIN_C, false><<<NSITES / 64, 128, 0, stream>>>(
        featsH, (const uint4*)Wss, nullptr, nullptr, ysc, sS, qS, 1);

    bn_finalize<<<1, 128, 0, stream>>>(s1, q1, g1, b1, a1, c1);
    bn_finalize<<<1, 128, 0, stream>>>(sS, qS, gsc, bsc, aS, cS);
    bn_relu_f16<<<NSITES * COUT_C / 256, 256, 0, stream>>>(y1, a1, c1, hH);

    // conv2 (128 -> 128, gather); y1 is dead, reuse as y2.
    spconv_wmma<COUT_C, true><<<NSITES / 64, 128, 0, stream>>>(
        hH, (const uint4*)W2s, in_idx, out_idx, y1, s2, q2, KOFF);
    bn_finalize<<<1, 128, 0, stream>>>(s2, q2, g2, b2, a2, c2);

    final_out<<<NSITES * COUT_C / 256, 256, 0, stream>>>(
        y1, a2, c2, ysc, aS, cS, out);
}
